// StreamingTransformerDecoder_23124103921872
// MI455X (gfx1250) — compile-verified, matrix operand order checked by reading, not run on hardware
//
#include <hip/hip_runtime.h>

// ---------------------------------------------------------------------------
// CDNA5 (gfx1250) streaming transformer decoder.
// All large GEMMs (day proj, patch embed, qkv, attention QK^T and AV, out
// proj, fc1/fc2, diphone head) run on v_wmma_f32_16x16x32_bf16 (wave32 WMMA).
// Weights converted once per call to bf16 [N,K] so A and B fragments are
// contiguous 16B loads matching the ISA 16-bit matrix VGPR layout.
// ---------------------------------------------------------------------------

typedef __bf16 bf16_t;
typedef __attribute__((ext_vector_type(8)))  bf16_t v8bf;
typedef __attribute__((ext_vector_type(16))) bf16_t v16bf;
typedef __attribute__((ext_vector_type(8)))  float  v8f;

#define BB     8
#define TT     8192
#define TM     8193          // conv output length
#define NC     256           // NEURAL
#define KLEN_  14
#define SS     585           // npatch
#define SPAD   640           // keys padded to multiple of 64
#define MTOK   (BB * SS)     // 4680 tokens
#define DD     1024
#define HH     16
#define HD     64
#define FFD    4096
#define LL     8
#define PEIN   3584
#define NBLANK 41
#define NDIP   40

// ===========================================================================
// Batched bf16 WMMA GEMM: C[z][M,N] (f32, row stride ldC)
//   = A[z][M,K] (bf16 row-major) x Bt[z][N,K] (bf16 row-major == B^T)
// wave -> 16x64 tile (4 accumulators), block = 8 waves -> 128x64 tile.
// N multiple of 64, K multiple of 32; M arbitrary (clamped loads, guarded
// stores) so EXEC stays all-ones across the WMMA ops.
// ===========================================================================
__global__ void __launch_bounds__(256)
gemm_bf16_wmma(const bf16_t* __restrict__ A, const bf16_t* __restrict__ Bt,
               float* __restrict__ C, int M, int N, int K, int ldC,
               long long sA, long long sB, long long sC) {
  const int z = blockIdx.z;
  A  += (long long)z * sA;
  Bt += (long long)z * sB;
  C  += (long long)z * sC;
  const int lane = threadIdx.x & 31;
  const int wave = threadIdx.x >> 5;
  const int l15  = lane & 15;
  const int kh   = lane >> 4;               // which K-half this lane holds
  const int mBase = (blockIdx.y * 8 + wave) * 16;
  const int nBase = blockIdx.x * 64;
  if (mBase >= M) return;                   // wave-uniform

  int aRow = mBase + l15; if (aRow > M - 1) aRow = M - 1;
  const bf16_t* ap  = A  + (size_t)aRow * K;
  const bf16_t* bp0 = Bt + (size_t)(nBase + l15) * K;
  const bf16_t* bp1 = bp0 + (size_t)16 * K;
  const bf16_t* bp2 = bp0 + (size_t)32 * K;
  const bf16_t* bp3 = bp0 + (size_t)48 * K;

  v8f acc[4] = {};
  const int koff0 = kh * 8;                 // ISA 16-bit A/B layout: lanes 0-15
  const int koff1 = 16 + kh * 8;            // hold K 0..7 & 16..23, lanes 16-31
                                            // hold K 8..15 & 24..31
  union U { v16bf v; v8bf h[2]; };

  for (int k0 = 0; k0 < K; k0 += 32) {
    U a, b0, b1, b2, b3;
    a.h[0]  = *(const v8bf*)(ap  + k0 + koff0);
    a.h[1]  = *(const v8bf*)(ap  + k0 + koff1);
    b0.h[0] = *(const v8bf*)(bp0 + k0 + koff0);
    b0.h[1] = *(const v8bf*)(bp0 + k0 + koff1);
    b1.h[0] = *(const v8bf*)(bp1 + k0 + koff0);
    b1.h[1] = *(const v8bf*)(bp1 + k0 + koff1);
    b2.h[0] = *(const v8bf*)(bp2 + k0 + koff0);
    b2.h[1] = *(const v8bf*)(bp2 + k0 + koff1);
    b3.h[0] = *(const v8bf*)(bp3 + k0 + koff0);
    b3.h[1] = *(const v8bf*)(bp3 + k0 + koff1);
    if (k0 + 32 < K) {                      // prefetch next K slab
      __builtin_prefetch(ap  + k0 + 32, 0, 1);
      __builtin_prefetch(bp0 + k0 + 32, 0, 1);
      __builtin_prefetch(bp1 + k0 + 32, 0, 1);
      __builtin_prefetch(bp2 + k0 + 32, 0, 1);
      __builtin_prefetch(bp3 + k0 + 32, 0, 1);
    }
    acc[0] = __builtin_amdgcn_wmma_f32_16x16x32_bf16(false, a.v, false, b0.v,
                                                     (short)0, acc[0], false, false);
    acc[1] = __builtin_amdgcn_wmma_f32_16x16x32_bf16(false, a.v, false, b1.v,
                                                     (short)0, acc[1], false, false);
    acc[2] = __builtin_amdgcn_wmma_f32_16x16x32_bf16(false, a.v, false, b2.v,
                                                     (short)0, acc[2], false, false);
    acc[3] = __builtin_amdgcn_wmma_f32_16x16x32_bf16(false, a.v, false, b3.v,
                                                     (short)0, acc[3], false, false);
  }

  // C/D layout: lane holds column n = l15; VGPR j -> row (kh*8 + j)
#pragma unroll
  for (int nt = 0; nt < 4; ++nt) {
    const int col = nBase + nt * 16 + l15;
#pragma unroll
    for (int j = 0; j < 8; ++j) {
      const int row = mBase + kh * 8 + j;
      if (row < M) C[(size_t)row * ldC + col] = acc[nt][j];
    }
  }
}

// ===========================================================================
// Weight prep: f32 [K,N] -> bf16 [N,K] (transposed), LDS-tiled.
// ===========================================================================
__global__ void transpose_to_bf16_kernel(const float* __restrict__ w,
                                         bf16_t* __restrict__ wt, int K, int N) {
  __shared__ float tile[32][33];
  const int k0 = blockIdx.y * 32, n0 = blockIdx.x * 32;
  for (int i = threadIdx.y; i < 32; i += 8) {
    int k = k0 + i, n = n0 + threadIdx.x;
    tile[i][threadIdx.x] = (k < K && n < N) ? w[(size_t)k * N + n] : 0.f;
  }
  __syncthreads();
  for (int i = threadIdx.y; i < 32; i += 8) {
    int n = n0 + i, k = k0 + threadIdx.x;
    if (n < N && k < K) wt[(size_t)n * K + k] = (bf16_t)tile[threadIdx.x][i];
  }
}

// day_W[day_idx[b]] -> bf16 transposed per batch: w_day[b][n][k]
__global__ void day_prep_kernel(const float* __restrict__ dayW,
                                const int* __restrict__ day_idx,
                                bf16_t* __restrict__ wt) {
  long long idx = (long long)blockIdx.x * 256 + threadIdx.x;
  if (idx >= 8LL * 256 * 256) return;
  int b = (int)(idx >> 16);
  int rem = (int)(idx & 65535);
  int n = rem >> 8, k = rem & 255;
  wt[idx] = (bf16_t)dayW[((long long)day_idx[b] * 256 + k) * 256 + n];
}

// ===========================================================================
// Depthwise gaussian smoothing along time (kernel 20, pad 10/10 -> len 8193).
// ===========================================================================
__global__ void gauss_conv_kernel(const float* __restrict__ x,
                                  float* __restrict__ y) {
  long long idx = (long long)blockIdx.x * 256 + threadIdx.x;
  if (idx >= 8LL * TM * NC) return;
  int c = (int)(idx % NC);
  long long r = idx / NC;
  int t = (int)(r % TM);
  int b = (int)(r / TM);
  float kc[20]; float ssum = 0.f;
  for (int j = 0; j < 20; ++j) {
    float z = ((float)j - 9.5f) * 0.5f;     // (j - mean)/sigma, sigma=2
    kc[j] = __expf(-0.5f * z * z);
    ssum += kc[j];
  }
  const float inv = 1.f / ssum;
  const float* xb = x + (long long)b * TT * NC + c;
  float acc = 0.f;
  for (int j = 0; j < 20; ++j) {
    int ti = t + j - 10;
    if (ti >= 0 && ti < TT) acc += kc[j] * xb[(long long)ti * NC];
  }
  y[idx] = acc * inv;
}

__global__ void f32_to_bf16_kernel(const float* __restrict__ in,
                                   bf16_t* __restrict__ out, long long n) {
  long long idx = (long long)blockIdx.x * 256 + threadIdx.x;
  if (idx < n) out[idx] = (bf16_t)in[idx];
}

// day bias + softsign + length mask (in place on day-GEMM output)
__global__ void day_epilogue_kernel(float* __restrict__ c,
                                    const float* __restrict__ day_b,
                                    const int* __restrict__ day_idx,
                                    const int* __restrict__ lengths) {
  long long idx = (long long)blockIdx.x * 256 + threadIdx.x;
  if (idx >= 8LL * TM * NC) return;
  int ch = (int)(idx % NC);
  long long r = idx / NC;
  int t = (int)(r % TM);
  int b = (int)(r / TM);
  float v = c[idx] + day_b[(long long)day_idx[b] * NC + ch];
  v = v / (1.f + fabsf(v));
  if (t >= lengths[b]) v = 0.f;
  c[idx] = v;
}

// ===========================================================================
// LayerNorm (two-pass, matches reference), fused f32->bf16 for next GEMM A.
// Input row r maps to in + (r/S)*bStride + (r%S)*rStride  (flexible gather:
// the patch-embed input is a contiguous 3584-float slice of the day output).
// ===========================================================================
__global__ void __launch_bounds__(256)
ln_kernel(const float* __restrict__ in, const float* __restrict__ w,
          const float* __restrict__ bb, bf16_t* __restrict__ obf,
          float* __restrict__ of32, int Dn, int S,
          long long bStride, long long rStride, float eps) {
  const int r = blockIdx.x, tid = threadIdx.x;
  const float* x = in + (long long)(r / S) * bStride + (long long)(r % S) * rStride;
  __shared__ float red[256];
  float s = 0.f;
  for (int i = tid; i < Dn; i += 256) s += x[i];
  red[tid] = s; __syncthreads();
  for (int o = 128; o > 0; o >>= 1) { if (tid < o) red[tid] += red[tid + o]; __syncthreads(); }
  const float mu = red[0] / (float)Dn;
  __syncthreads();
  float v = 0.f;
  for (int i = tid; i < Dn; i += 256) { float d = x[i] - mu; v += d * d; }
  red[tid] = v; __syncthreads();
  for (int o = 128; o > 0; o >>= 1) { if (tid < o) red[tid] += red[tid + o]; __syncthreads(); }
  const float rs = rsqrtf(red[0] / (float)Dn + eps);
  for (int i = tid; i < Dn; i += 256) {
    float y = (x[i] - mu) * rs * w[i] + bb[i];
    if (obf)  obf[(long long)r * Dn + i] = (bf16_t)y;
    if (of32) of32[(long long)r * Dn + i] = y;
  }
}

__global__ void add_bias_kernel(float* __restrict__ c, const float* __restrict__ bias,
                                long long n, int N) {
  long long idx = (long long)blockIdx.x * 256 + threadIdx.x;
  if (idx < n) c[idx] += bias[idx % N];
}

// h += (c + bias), optionally zeroed at padded query rows
__global__ void resid_add_kernel(float* __restrict__ h, const float* __restrict__ c,
                                 const float* __restrict__ bias,
                                 const int* __restrict__ lengths,
                                 int useMask, long long n) {
  long long idx = (long long)blockIdx.x * 256 + threadIdx.x;
  if (idx >= n) return;
  int d = (int)(idx % DD);
  long long r = idx / DD;
  float v = c[idx] + bias[d];
  if (useMask) {
    int b = (int)(r / SS), s = (int)(r % SS);
    int eff = lengths[b] / KLEN_; if (eff < 1) eff = 1;
    if (s >= eff) v = 0.f;
  }
  h[idx] += v;
}

// bias + exact GELU, fused f32->bf16 for fc2 A operand
__global__ void bias_gelu_bf16_kernel(const float* __restrict__ c,
                                      const float* __restrict__ bias,
                                      bf16_t* __restrict__ o, long long n, int N) {
  long long idx = (long long)blockIdx.x * 256 + threadIdx.x;
  if (idx >= n) return;
  float x = c[idx] + bias[idx % N];
  o[idx] = (bf16_t)(0.5f * x * (1.f + erff(x * 0.70710678118654752f)));
}

// ===========================================================================
// Attention operand prep: qkv f32 [tok,3072] (+bias, fused) ->
//   Q  bf16 [bh][585][64]   (A of QK^T)
//   K  bf16 [bh][640][64]   (Bt of QK^T; keys 585..639 zero, masked later)
//   Vt bf16 [bh][64][640]   (Bt of P.V; columns 585..639 zero)
// ===========================================================================
__global__ void qkv_split_kernel(const float* __restrict__ qkv,
                                 const float* __restrict__ qkvb,
                                 bf16_t* __restrict__ qb,
                                 bf16_t* __restrict__ kb,
                                 bf16_t* __restrict__ vtb) {
  long long idx = (long long)blockIdx.x * 256 + threadIdx.x;
  if (idx >= (long long)BB * HH * SPAD * HD) return;
  int d = (int)(idx % HD);
  long long r = idx / HD;
  int s = (int)(r % SPAD);
  int bh = (int)(r / SPAD);
  int b = bh >> 4, h = bh & 15;
  float qv = 0.f, kv = 0.f, vv = 0.f;
  if (s < SS) {
    const float* base = qkv + (long long)(b * SS + s) * (3 * DD) + h * HD + d;
    qv = base[0]      + qkvb[h * HD + d];
    kv = base[DD]     + qkvb[DD + h * HD + d];
    vv = base[2 * DD] + qkvb[2 * DD + h * HD + d];
    qb[((long long)bh * SS + s) * HD + d] = (bf16_t)qv;
  }
  kb[((long long)bh * SPAD + s) * HD + d] = (bf16_t)kv;
  vtb[((long long)bh * HD + d) * SPAD + s] = (bf16_t)vv;
}

// scale + causal(-1e9 additive) + pad(-1e30 overwrite) + softmax -> bf16 probs
__global__ void __launch_bounds__(256)
attn_softmax_kernel(const float* __restrict__ sbuf,
                    const int* __restrict__ lengths,
                    bf16_t* __restrict__ pbf) {
  const int row = blockIdx.x;                 // bh*585 + q
  const int q = row % SS;
  const int bh = row / SS;
  const int b = bh >> 4;
  const int tid = threadIdx.x;
  int eff = lengths[b] / KLEN_; if (eff < 1) eff = 1;
  const float* srow = sbuf + (long long)row * SPAD;
  bf16_t* prow = pbf + (long long)row * SPAD;
  __shared__ float red[256];
  float m = -1e30f;
  float lg[3];                                // 640/256 -> up to 3 per thread
#pragma unroll
  for (int it = 0; it < 3; ++it) {
    int kk = tid + it * 256;
    float sc = -1e30f;
    if (kk < SS && kk < eff) {
      sc = srow[kk] * 0.125f;                 // 1/sqrt(64)
      if (kk > q) sc += -1e9f;                // additive causal mask
    }
    lg[it] = sc;
    if (kk < SPAD) m = fmaxf(m, sc);
  }
  red[tid] = m; __syncthreads();
  for (int o = 128; o > 0; o >>= 1) { if (tid < o) red[tid] = fmaxf(red[tid], red[tid + o]); __syncthreads(); }
  m = red[0]; __syncthreads();
  float ss = 0.f;
#pragma unroll
  for (int it = 0; it < 3; ++it) {
    int kk = tid + it * 256;
    if (kk < SPAD) ss += __expf(lg[it] - m);
  }
  red[tid] = ss; __syncthreads();
  for (int o = 128; o > 0; o >>= 1) { if (tid < o) red[tid] += red[tid + o]; __syncthreads(); }
  const float inv = 1.f / red[0];
#pragma unroll
  for (int it = 0; it < 3; ++it) {
    int kk = tid + it * 256;
    if (kk < SPAD) prow[kk] = (bf16_t)(__expf(lg[it] - m) * inv);
  }
}

// per-head AV result [bh][585][64] f32 -> token-major bf16 [tok][1024]
__global__ void av_merge_bf16_kernel(const float* __restrict__ av,
                                     bf16_t* __restrict__ out) {
  long long idx = (long long)blockIdx.x * 256 + threadIdx.x;
  if (idx >= (long long)BB * HH * SS * HD) return;
  int d = (int)(idx % HD);
  long long r = idx / HD;
  int s = (int)(r % SS);
  int bh = (int)(r / SS);
  int b = bh >> 4, h = bh & 15;
  out[((long long)(b * SS + s)) * DD + h * HD + d] = (bf16_t)av[idx];
}

// ===========================================================================
// 41-wide classifier head with log_softmax (+ optional exp() output).
// ===========================================================================
__global__ void __launch_bounds__(64)
head41_kernel(const float* __restrict__ A, const float* __restrict__ W,
              const float* __restrict__ bias, float* __restrict__ lp,
              float* __restrict__ eout, int K) {
  __shared__ float xs[DD];
  __shared__ float lg[NBLANK];
  __shared__ float st[2];
  const int r = blockIdx.x, tid = threadIdx.x;
  const float* x = A + (long long)r * K;
  for (int i = tid; i < K; i += 64) xs[i] = x[i];
  __syncthreads();
  if (tid < NBLANK) {
    float acc = bias[tid];
    for (int k = 0; k < K; ++k) acc += xs[k] * W[(long long)k * NBLANK + tid];
    lg[tid] = acc;
  }
  __syncthreads();
  if (tid == 0) {
    float m = -1e30f;
    for (int n = 0; n < NBLANK; ++n) m = fmaxf(m, lg[n]);
    float s = 0.f;
    for (int n = 0; n < NBLANK; ++n) s += __expf(lg[n] - m);
    st[0] = m; st[1] = __logf(s);
  }
  __syncthreads();
  if (tid < NBLANK) {
    float v = lg[tid] - st[0] - st[1];
    lp[(long long)r * NBLANK + tid] = v;
    if (eout) eout[(long long)r * NBLANK + tid] = __expf(v);
  }
}

// h += exp(inter_lp) @ fb_W + fb_b   (K = 41)
__global__ void feedback_kernel(float* __restrict__ h, const float* __restrict__ e,
                                const float* __restrict__ fbW,
                                const float* __restrict__ fbB, long long n) {
  long long idx = (long long)blockIdx.x * 256 + threadIdx.x;
  if (idx >= n) return;
  int d = (int)(idx % DD);
  long long r = idx / DD;
  const float* er = e + r * NBLANK;
  float acc = fbB[d];
  for (int nn = 0; nn < NBLANK; ++nn) acc += er[nn] * fbW[(long long)nn * DD + d];
  h[idx] += acc;
}

// ===========================================================================
// Diphone marginalization: per row of 1600 logits -> log_softmax ->
// left/right LSE over the 40x40 grid -> logaddexp - ln2 -> renormalize.
// ===========================================================================
__global__ void __launch_bounds__(256)
diphone_kernel(const float* __restrict__ C, const float* __restrict__ bias,
               float* __restrict__ out) {
  __shared__ float v[NDIP * NDIP];
  __shared__ float red[256];
  __shared__ float lr[2 * NDIP];
  __shared__ float st[1];
  const int r = blockIdx.x, tid = threadIdx.x;
  const float* x = C + (long long)r * (NDIP * NDIP);
  float lm = -1e30f;
  for (int i = tid; i < NDIP * NDIP; i += 256) {
    float t = x[i] + bias[i]; v[i] = t; lm = fmaxf(lm, t);
  }
  red[tid] = lm; __syncthreads();
  for (int o = 128; o > 0; o >>= 1) { if (tid < o) red[tid] = fmaxf(red[tid], red[tid + o]); __syncthreads(); }
  const float m = red[0]; __syncthreads();
  float ls = 0.f;
  for (int i = tid; i < NDIP * NDIP; i += 256) ls += __expf(v[i] - m);
  red[tid] = ls; __syncthreads();
  for (int o = 128; o > 0; o >>= 1) { if (tid < o) red[tid] += red[tid + o]; __syncthreads(); }
  const float logZ = m + __logf(red[0]);
  __syncthreads();
  for (int i = tid; i < NDIP * NDIP; i += 256) v[i] -= logZ;
  __syncthreads();
  if (tid < NDIP) {                                  // left: LSE over columns
    float mm = -1e30f;
    for (int j = 0; j < NDIP; ++j) mm = fmaxf(mm, v[tid * NDIP + j]);
    float ss = 0.f;
    for (int j = 0; j < NDIP; ++j) ss += __expf(v[tid * NDIP + j] - mm);
    lr[tid] = mm + __logf(ss);
  } else if (tid < 2 * NDIP) {                       // right: LSE over rows
    const int j = tid - NDIP;
    float mm = -1e30f;
    for (int i2 = 0; i2 < NDIP; ++i2) mm = fmaxf(mm, v[i2 * NDIP + j]);
    float ss = 0.f;
    for (int i2 = 0; i2 < NDIP; ++i2) ss += __expf(v[i2 * NDIP + j] - mm);
    lr[tid] = mm + __logf(ss);
  }
  __syncthreads();
  if (tid < NDIP) {
    float a = lr[tid], b = lr[tid + NDIP];
    float mx = fmaxf(a, b);
    lr[tid] = mx + __logf(__expf(a - mx) + __expf(b - mx)) - 0.69314718055994531f;
  }
  __syncthreads();
  if (tid == 0) {
    float mm = -1e30f;
    for (int i2 = 0; i2 < NDIP; ++i2) mm = fmaxf(mm, lr[i2]);
    float ss = 0.f;
    for (int i2 = 0; i2 < NDIP; ++i2) ss += __expf(lr[i2] - mm);
    st[0] = mm + __logf(ss);
  }
  __syncthreads();
  if (tid < NDIP) out[(long long)r * NDIP + tid] = lr[tid] - st[0];
}

// ===========================================================================
// Host orchestration
// ===========================================================================
static inline long long ceil_div_ll(long long a, long long b) { return (a + b - 1) / b; }

extern "C" void kernel_launch(void* const* d_in, const int* in_sizes, int n_in,
                              void* d_out, int out_size, void* d_ws, size_t ws_size,
                              hipStream_t stream) {
  (void)in_sizes; (void)n_in; (void)out_size; (void)ws_size;
  // ---- inputs (setup_inputs order) ----
  const float* neural     = (const float*)d_in[0];
  const int*   lengths    = (const int*)  d_in[1];
  const int*   day_idx    = (const int*)  d_in[2];
  const float* day_W      = (const float*)d_in[3];
  const float* day_b      = (const float*)d_in[4];
  const float* pe_ln1_w   = (const float*)d_in[5];
  const float* pe_ln1_b   = (const float*)d_in[6];
  const float* pe_W       = (const float*)d_in[7];
  const float* pe_b       = (const float*)d_in[8];
  const float* pe_ln2_w   = (const float*)d_in[9];
  const float* pe_ln2_b   = (const float*)d_in[10];
  const float* ln_attn_w  = (const float*)d_in[11];
  const float* ln_attn_b  = (const float*)d_in[12];
  const float* qkv_W      = (const float*)d_in[13];
  const float* qkv_b      = (const float*)d_in[14];
  const float* ao_W       = (const float*)d_in[15];
  const float* ao_b       = (const float*)d_in[16];
  const float* ff_ln_w    = (const float*)d_in[17];
  const float* ff_ln_b    = (const float*)d_in[18];
  const float* fc1_W      = (const float*)d_in[19];
  const float* fc1_b      = (const float*)d_in[20];
  const float* fc2_W      = (const float*)d_in[21];
  const float* fc2_b      = (const float*)d_in[22];
  const float* final_ln_w = (const float*)d_in[23];
  const float* final_ln_b = (const float*)d_in[24];
  const float* ih_W       = (const float*)d_in[25];
  const float* ih_b       = (const float*)d_in[26];
  const float* fb_W       = (const float*)d_in[27];
  const float* fb_b       = (const float*)d_in[28];
  const float* fh_W       = (const float*)d_in[29];
  const float* fh_b       = (const float*)d_in[30];
  const float* dh_W       = (const float*)d_in[31];
  const float* dh_b       = (const float*)d_in[32];
  float* outF = (float*)d_out;

  // ---- workspace layout ----
  char* ws = (char*)d_ws;
  size_t off = 0;
  auto alloc = [&](size_t bytes) -> char* {
    char* p = ws + off;
    off += (bytes + 255) & ~(size_t)255;
    return p;
  };
  bf16_t* w_day = (bf16_t*)alloc(8LL * 256 * 256 * 2);
  bf16_t* w_pe  = (bf16_t*)alloc((size_t)DD * PEIN * 2);
  bf16_t* w_qkv = (bf16_t*)alloc((size_t)LL * 3 * DD * DD * 2);
  bf16_t* w_ao  = (bf16_t*)alloc((size_t)LL * DD * DD * 2);
  bf16_t* w_fc1 = (bf16_t*)alloc((size_t)LL * FFD * DD * 2);
  bf16_t* w_fc2 = (bf16_t*)alloc((size_t)LL * DD * FFD * 2);
  bf16_t* w_dh  = (bf16_t*)alloc((size_t)NDIP * NDIP * DD * 2);
  float*  xs    = (float*) alloc(8LL * TM * NC * 4);                  // conv out
  bf16_t* bfA   = (bf16_t*)alloc((size_t)MTOK * FFD * 2);             // shared A buf
  float*  cbuf  = (float*) alloc((size_t)MTOK * FFD * 4);             // shared C buf
  float*  hbuf  = (float*) alloc((size_t)MTOK * DD * 4);              // residual
  float*  obuf  = (float*) alloc((size_t)MTOK * DD * 4);              // attn/ff tmp
  bf16_t* qh    = (bf16_t*)alloc((size_t)BB * HH * SS * HD * 2);      // Q per head
  bf16_t* kh    = (bf16_t*)alloc((size_t)BB * HH * SPAD * HD * 2);    // K per head
  bf16_t* vt    = (bf16_t*)alloc((size_t)BB * HH * HD * SPAD * 2);    // V^T per head
  float*  sbuf  = (float*) alloc((size_t)BB * HH * SS * SPAD * 4);    // attn logits
  bf16_t* pbf   = (bf16_t*)alloc((size_t)BB * HH * SS * SPAD * 2);    // bf16 probs
  float*  ebuf  = (float*) alloc((size_t)MTOK * NBLANK * 4);          // exp(inter)

  auto ew = [&](long long n) { return dim3((unsigned)ceil_div_ll(n, 256)); };
  auto tr = [&](const float* w, bf16_t* wt, int K, int N) {
    transpose_to_bf16_kernel<<<dim3((N + 31) / 32, (K + 31) / 32), dim3(32, 8), 0, stream>>>(w, wt, K, N);
  };
  auto gemm = [&](const bf16_t* A, const bf16_t* Bt, float* C, int M, int N, int K) {
    gemm_bf16_wmma<<<dim3(N / 64, (unsigned)ceil_div_ll(M, 128), 1), 256, 0, stream>>>(
        A, Bt, C, M, N, K, N, 0, 0, 0);
  };
  auto gemmB = [&](const bf16_t* A, const bf16_t* Bt, float* C, int M, int N, int K,
                   int ldC, int batch, long long sA, long long sB, long long sC) {
    gemm_bf16_wmma<<<dim3(N / 64, (unsigned)ceil_div_ll(M, 128), batch), 256, 0, stream>>>(
        A, Bt, C, M, N, K, ldC, sA, sB, sC);
  };

  // ---- phase A: weight conversion (bf16, transposed [N,K]) ----
  tr(pe_W, w_pe, PEIN, DD);
  for (int l = 0; l < LL; ++l) {
    tr(qkv_W + (size_t)l * DD * 3 * DD, w_qkv + (size_t)l * 3 * DD * DD, DD, 3 * DD);
    tr(ao_W  + (size_t)l * DD * DD,     w_ao  + (size_t)l * DD * DD,     DD, DD);
    tr(fc1_W + (size_t)l * DD * FFD,    w_fc1 + (size_t)l * FFD * DD,    DD, FFD);
    tr(fc2_W + (size_t)l * FFD * DD,    w_fc2 + (size_t)l * DD * FFD,    FFD, DD);
  }
  tr(dh_W, w_dh, DD, NDIP * NDIP);
  day_prep_kernel<<<ew(8LL * 256 * 256), 256, 0, stream>>>(day_W, day_idx, w_day);

  // ---- phase B: smoothing + day linear + patch embed ----
  const long long nConv = 8LL * TM * NC;
  gauss_conv_kernel<<<ew(nConv), 256, 0, stream>>>(neural, xs);
  f32_to_bf16_kernel<<<ew(nConv), 256, 0, stream>>>(xs, bfA, nConv);
  // one GEMM per batch element (per-day weights)
  for (int b = 0; b < BB; ++b)
    gemm(bfA + (size_t)b * TM * NC, w_day + (size_t)b * 256 * 256,
         cbuf + (size_t)b * TM * NC, TM, NC, NC);
  day_epilogue_kernel<<<ew(nConv), 256, 0, stream>>>(cbuf, day_b, day_idx, lengths);

  // patch LN over contiguous [14*256] slices of the day output
  ln_kernel<<<MTOK, 256, 0, stream>>>(cbuf, pe_ln1_w, pe_ln1_b, bfA, nullptr,
                                      PEIN, SS, (long long)TM * NC, (long long)KLEN_ * NC, 1e-6f);
  gemm(bfA, w_pe, cbuf, MTOK, DD, PEIN);
  add_bias_kernel<<<ew((long long)MTOK * DD), 256, 0, stream>>>(cbuf, pe_b, (long long)MTOK * DD, DD);
  ln_kernel<<<MTOK, 256, 0, stream>>>(cbuf, pe_ln2_w, pe_ln2_b, nullptr, hbuf,
                                      DD, MTOK, 0, DD, 1e-6f);

  // ---- phase C: transformer layers ----
  const long long sQ = (long long)SS * HD;      // per-head Q stride
  const long long sK = (long long)SPAD * HD;    // per-head K stride
  const long long sV = (long long)HD * SPAD;    // per-head V^T stride
  const long long sS = (long long)SS * SPAD;    // per-head logits/probs stride
  const long long sO = (long long)SS * HD;      // per-head AV stride
  for (int l = 0; l < LL; ++l) {
    ln_kernel<<<MTOK, 256, 0, stream>>>(hbuf, ln_attn_w + (size_t)l * DD,
                                        ln_attn_b + (size_t)l * DD, bfA, nullptr,
                                        DD, MTOK, 0, DD, 1e-5f);
    gemm(bfA, w_qkv + (size_t)l * 3 * DD * DD, cbuf, MTOK, 3 * DD, DD);
    qkv_split_kernel<<<ew((long long)BB * HH * SPAD * HD), 256, 0, stream>>>(
        cbuf, qkv_b + (size_t)l * 3 * DD, qh, kh, vt);
    // scores[bh][585][640] = Q.K^T  (128 batched WMMA GEMMs in one launch)
    gemmB(qh, kh, sbuf, SS, SPAD, HD, SPAD, BB * HH, sQ, sK, sS);
    attn_softmax_kernel<<<BB * HH * SS, 256, 0, stream>>>(sbuf, lengths, pbf);
    // AV[bh][585][64] = P.V  (K = 640, padded keys have zero prob/value)
    gemmB(pbf, vt, obuf, SS, HD, SPAD, HD, BB * HH, sS, sV, sO);
    av_merge_bf16_kernel<<<ew((long long)MTOK * DD), 256, 0, stream>>>(obuf, bfA);
    gemm(bfA, w_ao + (size_t)l * DD * DD, obuf, MTOK, DD, DD);
    resid_add_kernel<<<ew((long long)MTOK * DD), 256, 0, stream>>>(
        hbuf, obuf, ao_b + (size_t)l * DD, lengths, 1, (long long)MTOK * DD);

    ln_kernel<<<MTOK, 256, 0, stream>>>(hbuf, ff_ln_w + (size_t)l * DD,
                                        ff_ln_b + (size_t)l * DD, bfA, nullptr,
                                        DD, MTOK, 0, DD, 1e-5f);
    gemm(bfA, w_fc1 + (size_t)l * FFD * DD, cbuf, MTOK, FFD, DD);
    bias_gelu_bf16_kernel<<<ew((long long)MTOK * FFD), 256, 0, stream>>>(
        cbuf, fc1_b + (size_t)l * FFD, bfA, (long long)MTOK * FFD, FFD);
    gemm(bfA, w_fc2 + (size_t)l * DD * FFD, obuf, MTOK, DD, FFD);
    resid_add_kernel<<<ew((long long)MTOK * DD), 256, 0, stream>>>(
        hbuf, obuf, fc2_b + (size_t)l * DD, lengths, 0, (long long)MTOK * DD);

    if (l == 3) {  // INTER head + feedback
      head41_kernel<<<MTOK, 64, 0, stream>>>(hbuf, ih_W, ih_b,
                                             outF + (size_t)MTOK * NBLANK, ebuf, DD);
      feedback_kernel<<<ew((long long)MTOK * DD), 256, 0, stream>>>(
          hbuf, ebuf, fb_W, fb_b, (long long)MTOK * DD);
    }
  }

  // ---- phase D: final LN + heads ----
  ln_kernel<<<MTOK, 256, 0, stream>>>(hbuf, final_ln_w, final_ln_b, bfA, obuf,
                                      DD, MTOK, 0, DD, 1e-5f);
  head41_kernel<<<MTOK, 64, 0, stream>>>(obuf, fh_W, fh_b, outF, nullptr, DD);
  gemm(bfA, w_dh, cbuf, MTOK, NDIP * NDIP, DD);
  diphone_kernel<<<MTOK, 256, 0, stream>>>(cbuf, dh_b,
                                           outF + 2 * (size_t)MTOK * NBLANK);
}
